// graph_structure_learner_36103495090409
// MI455X (gfx1250) — matrix operand order repaired and checked by reading, not compiled, not run on hardware
//
#include <hip/hip_runtime.h>

typedef __attribute__((ext_vector_type(16))) _Float16 v16h;
typedef __attribute__((ext_vector_type(8)))  float    v8f;

#define LAMDA  0.5f
#define BN_EPS 1e-5f
#define SLOPE  0.01f
#define EMB    32
#define HID    64

__global__ void gsl_zero_flags(unsigned char* flags, int n) {
  int i = blockIdx.x * blockDim.x + threadIdx.x;
  if (i < n) flags[i] = 0;
}

__global__ void gsl_set_flags(const int* __restrict__ ori, unsigned char* flags, int n) {
  int i = blockIdx.x * blockDim.x + threadIdx.x;
  if (i < n) flags[ori[i]] = 1;   // duplicates write same value: benign
}

// Persistent-wave kernel. One wave (32 lanes, wave32) owns a 16-edge tile per
// loop iteration. A: 16x64 f16 feature matrix, B: W0 64x64 f16 (VGPR-resident
// across all tiles), C/D f32. K=64 -> two chained v_wmma_f32_16x16x32_f16 per
// N-tile; N=64 -> 4 N-tiles => 8 WMMAs per tile.
__global__ __launch_bounds__(256)
void gsl_wmma_kernel(const float* __restrict__ h,
                     const float* __restrict__ rel_emb,
                     const float* __restrict__ W0,
                     const float* __restrict__ b0,
                     const float* __restrict__ bn_gamma,
                     const float* __restrict__ bn_beta,
                     const float* __restrict__ bn_mean,
                     const float* __restrict__ bn_var,
                     const float* __restrict__ Ws,
                     const float* __restrict__ bs,
                     const int*   __restrict__ row,
                     const int*   __restrict__ col,
                     const int*   __restrict__ etype,
                     const unsigned char* __restrict__ flags,
                     float* __restrict__ out,
                     int E)
{
  const int lane = threadIdx.x & 31;
  const int wave = threadIdx.x >> 5;
  const int half = lane >> 4;      // which 16-lane half of the wave
  const int l15  = lane & 15;

  // ---- B fragments: W0[k][n] (row-major 64x64), resident for whole kernel ----
  // 16-bit B layout: element e of half h holds K=base+8h+e (e<8),
  // K=base+16+8h+(e-8) (e>=8); N = lane&15 within N-tile.
  v16h Bf[4][2];
  #pragma unroll
  for (int n = 0; n < 4; ++n) {
    const int N = n * 16 + l15;
    #pragma unroll
    for (int kt = 0; kt < 2; ++kt) {
      const int Kbase = kt * 32 + half * 8;
      #pragma unroll
      for (int e = 0; e < 16; ++e)
        Bf[n][kt][e] = (_Float16)W0[(Kbase + e + ((e >> 3) << 3)) * HID + N];
    }
  }

  // ---- fold Linear bias + BatchNorm(eval) into per-column (s, o); Ws per col
  float s_[4], o_[4], w_[4];
  #pragma unroll
  for (int n = 0; n < 4; ++n) {
    const int N = n * 16 + l15;
    const float sc = bn_gamma[N] * rsqrtf(bn_var[N] + BN_EPS);
    s_[n] = sc;
    o_[n] = (b0[N] - bn_mean[N]) * sc + bn_beta[N];
    w_[n] = Ws[N];
  }
  const float bsv = bs[0];

  const int numTiles   = (E + 15) >> 4;
  const int tileStride = gridDim.x * 8;

  // ---- persistent grid-stride loop over 16-edge tiles (wave-uniform) ----
  for (int tile = blockIdx.x * 8 + wave; tile < numTiles; tile += tileStride) {
    const int base = tile * 16;

    const int gm  = base + l15;
    const int gmc = gm < E ? gm : (E - 1);        // clamp tail lanes, stay active
    const int ri = row[gmc], ci = col[gmc], ti = etype[gmc];
    const float* __restrict__ hs = h       + (long)ri * EMB;  // L2-resident (6.4MB)
    const float* __restrict__ hd = h       + (long)ci * EMB;
    const float* __restrict__ re = rel_emb + (long)ti * EMB;

    // ---- build A fragments in registers per ISA 16-bit A 16x32 layout:
    // lane l owns row M=l&15; fragment pair v (elements 2v,2v+1) covers
    // consecutive kk, kk+1 with kk = (v/4)*16 + half*8 + 2*(v%4)  -> float2 loads
    v16h A0, A1;
    #pragma unroll
    for (int v = 0; v < 8; ++v) {
      const int kk = ((v >> 2) << 4) + (half << 3) + ((v & 3) << 1);
      const float2 a2 = *(const float2*)(hs + kk);
      const float2 b2 = *(const float2*)(hd + kk);
      const float2 r2 = *(const float2*)(re + kk);
      A0[2*v]     = (_Float16)__expf(-fabsf(a2.x - b2.x)); // feature k = kk
      A0[2*v + 1] = (_Float16)__expf(-fabsf(a2.y - b2.y)); // feature k = kk+1
      A1[2*v]     = (_Float16)r2.x;                        // feature k = 32+kk
      A1[2*v + 1] = (_Float16)r2.y;
    }

    // ---- 8 WMMAs: 4 N-tiles x 2 K-slices, chained f32 accumulation ----
    v8f acc[4] = {};
    #pragma unroll
    for (int n = 0; n < 4; ++n) {
      acc[n] = __builtin_amdgcn_wmma_f32_16x16x32_f16(
          false, A0, false, Bf[n][0], (short)0, acc[n], false, false);
      acc[n] = __builtin_amdgcn_wmma_f32_16x16x32_f16(
          false, A1, false, Bf[n][1], (short)0, acc[n], false, false);
    }

    // ---- epilogue: BN+LeakyReLU, dot with Ws across N (columns = lanes) ----
    // C/D layout: VGPR r holds M=r (lanes 0-15) / M=8+r (lanes 16-31), N=lane&15.
    float part[8];
    #pragma unroll
    for (int r = 0; r < 8; ++r) part[r] = 0.f;
    #pragma unroll
    for (int n = 0; n < 4; ++n) {
      const float sc = s_[n], of = o_[n], ww = w_[n];
      #pragma unroll
      for (int r = 0; r < 8; ++r) {
        float y = acc[n][r] * sc + of;
        y = (y >= 0.f) ? y : SLOPE * y;
        part[r] += y * ww;
      }
    }
    // butterfly sum over the 16 lanes of each half (xor<16 stays within half)
    #pragma unroll
    for (int r = 0; r < 8; ++r) {
      float t = part[r];
      t += __shfl_xor(t, 1, 32);
      t += __shfl_xor(t, 2, 32);
      t += __shfl_xor(t, 4, 32);
      t += __shfl_xor(t, 8, 32);
      part[r] = t;
    }

    if (l15 == 0) {
      #pragma unroll
      for (int r = 0; r < 8; ++r) {
        const int idx = base + half * 8 + r; // half 0 -> rows 0-7, half 1 -> 8-15
        if (idx < E) {
          float val = part[r] + bsv;
          if (flags[idx]) val = (1.f - LAMDA) * val + LAMDA;
          out[idx] = val;
        }
      }
    }
  }
}

extern "C" void kernel_launch(void* const* d_in, const int* in_sizes, int n_in,
                              void* d_out, int out_size, void* d_ws, size_t ws_size,
                              hipStream_t stream) {
  const float* h        = (const float*)d_in[0];
  const float* rel_emb  = (const float*)d_in[1];
  const float* W0       = (const float*)d_in[2];
  const float* b0       = (const float*)d_in[3];
  const float* bn_gamma = (const float*)d_in[4];
  const float* bn_beta  = (const float*)d_in[5];
  const float* bn_mean  = (const float*)d_in[6];
  const float* bn_var   = (const float*)d_in[7];
  const float* Ws       = (const float*)d_in[8];
  const float* bs       = (const float*)d_in[9];
  const int*   row      = (const int*)d_in[10];
  const int*   col      = (const int*)d_in[11];
  const int*   etype    = (const int*)d_in[12];
  const int*   ori_ids  = (const int*)d_in[13];

  const int E     = in_sizes[10];
  const int n_ori = in_sizes[13];

  unsigned char* flags = (unsigned char*)d_ws;  // E bytes of scratch
  float* out = (float*)d_out;

  gsl_zero_flags<<<(E + 511) / 512, 512, 0, stream>>>(flags, E);
  gsl_set_flags<<<(n_ori + 255) / 256, 256, 0, stream>>>(ori_ids, flags, n_ori);

  const int tiles    = (E + 15) / 16;
  int blocks = (tiles + 7) / 8;
  if (blocks > 4096) blocks = 4096;   // persistent waves, grid-stride over tiles
  gsl_wmma_kernel<<<blocks, 256, 0, stream>>>(
      h, rel_emb, W0, b0, bn_gamma, bn_beta, bn_mean, bn_var, Ws, bs,
      row, col, etype, flags, out, E);
}
